// BiGCN_34634616275006
// MI455X (gfx1250) — matrix operand order repaired and checked by reading, not compiled, not run on hardware
//
#include <hip/hip_runtime.h>
#include <hip/hip_bf16.h>

typedef __attribute__((ext_vector_type(16))) __bf16 v16bf;
typedef __attribute__((ext_vector_type(8)))  float  v8f;

#define IN_F  768
#define HID_F 64
#define OUT_F 64
#define NCLS  4

// ---------------------------------------------------------------- utilities
template<bool RELU>
__device__ __forceinline__ __bf16 cvbf(float v) {
    if (RELU) v = fmaxf(v, 0.0f);
    return (__bf16)v;
}

// Build a 16x32 bf16 A-fragment for one lane from row-major f32 data.
// Per ISA 7.12.2 (16-bit A 16x32): lane = (half, m); elements 0..7 hold
// K = 8*half + e, elements 8..15 hold K = 16 + 8*half + e.
// `ap` must already point at  row_base + 8*half  within the 32-wide K chunk.
template<bool RELU>
__device__ __forceinline__ v16bf make_afrag(const float* __restrict__ ap) {
    float4 q0 = *(const float4*)(ap);
    float4 q1 = *(const float4*)(ap + 4);
    float4 q2 = *(const float4*)(ap + 16);
    float4 q3 = *(const float4*)(ap + 20);
    v16bf af;
    af[0]  = cvbf<RELU>(q0.x); af[1]  = cvbf<RELU>(q0.y);
    af[2]  = cvbf<RELU>(q0.z); af[3]  = cvbf<RELU>(q0.w);
    af[4]  = cvbf<RELU>(q1.x); af[5]  = cvbf<RELU>(q1.y);
    af[6]  = cvbf<RELU>(q1.z); af[7]  = cvbf<RELU>(q1.w);
    af[8]  = cvbf<RELU>(q2.x); af[9]  = cvbf<RELU>(q2.y);
    af[10] = cvbf<RELU>(q2.z); af[11] = cvbf<RELU>(q2.w);
    af[12] = cvbf<RELU>(q3.x); af[13] = cvbf<RELU>(q3.y);
    af[14] = cvbf<RELU>(q3.z); af[15] = cvbf<RELU>(q3.w);
    return af;
}

// ---------------------------------------------------------------- degrees
__global__ void k_init_deg(float* __restrict__ dtd, float* __restrict__ dbu, int n) {
    int i = blockIdx.x * blockDim.x + threadIdx.x;
    if (i < n) { dtd[i] = 1.0f; dbu[i] = 1.0f; }   // self loop
}

__global__ void k_deg_edges(const int* __restrict__ dst, float* __restrict__ deg, int e) {
    int i = blockIdx.x * blockDim.x + threadIdx.x;
    if (i < e) atomicAdd(deg + dst[i], 1.0f);
}

__global__ void k_dinv(float* __restrict__ dtd, float* __restrict__ dbu, int n) {
    int i = blockIdx.x * blockDim.x + threadIdx.x;
    if (i < n) { dtd[i] = rsqrtf(dtd[i]); dbu[i] = rsqrtf(dbu[i]); }
}

// ---------------------------------------------------------------- weight packing
// B-fragment layout (32x16 bf16 B tile): lane L holds column n = L%16,
// K = 8*(L/16) + e for e < 8, K = 16 + 8*(L/16) + (e-8) for e >= 8.
// bf1 layout: [colTile c:8][kChunk kt:24][lane:32][e:16]
__global__ void k_pack_b1(const float* __restrict__ W1td, const float* __restrict__ W1bu,
                          __bf16* __restrict__ bf) {
    int t = blockIdx.x * blockDim.x + threadIdx.x;
    if (t >= 8 * 24 * 32 * 16) return;
    int e = t & 15;  int r = t >> 4;
    int L = r & 31;  r >>= 5;
    int kt = r % 24; int c = r / 24;
    int k = kt * 32 + ((L >> 4) << 3) + (e & 7) + ((e >> 3) << 4);
    int ncol = c * 16 + (L & 15);
    float v = (ncol < 64) ? W1td[k * 64 + ncol] : W1bu[k * 64 + (ncol - 64)];
    bf[t] = (__bf16)v;
}

// bf2 layout: [branch:2][colTile c:4][kChunk kt:2][lane:32][e:16] (K<64 rows of W2)
__global__ void k_pack_b2(const float* __restrict__ W2td, const float* __restrict__ W2bu,
                          __bf16* __restrict__ bf) {
    int t = blockIdx.x * blockDim.x + threadIdx.x;
    if (t >= 2 * 4 * 2 * 32 * 16) return;
    int e = t & 15;  int r = t >> 4;
    int L = r & 31;  r >>= 5;
    int kt = r & 1;  r >>= 1;
    int c  = r & 3;  r >>= 2;
    int branch = r;
    int k = kt * 32 + ((L >> 4) << 3) + (e & 7) + ((e >> 3) << 4);           // < 64 -> "h" rows of W2
    int ncol = c * 16 + (L & 15);
    const float* W2 = branch ? W2bu : W2td;
    bf[t] = (__bf16)W2[k * 64 + ncol];
}

// relu(root) @ W2[64:832]  -> rootproj [branch][B][64]
__global__ void k_rootproj(const float* __restrict__ root, const float* __restrict__ W2td,
                           const float* __restrict__ W2bu, float* __restrict__ rp, int nB) {
    int b = blockIdx.x, branch = blockIdx.y, j = threadIdx.x;  // 64 threads
    const float* W2 = branch ? W2bu : W2td;
    float acc = 0.0f;
    for (int k = 0; k < IN_F; ++k) {
        float rv = fmaxf(root[b * IN_F + k], 0.0f);
        acc += rv * W2[(HID_F + k) * OUT_F + j];
    }
    rp[(branch * nB + b) * 64 + j] = acc;
}

// ---------------------------------------------------------------- GEMM1: xw[N,128] = x @ [W1_td | W1_bu]  (bf16 WMMA, f32 accum)
__global__ __launch_bounds__(256)
void k_gemm1(const float* __restrict__ x, const __bf16* __restrict__ bf,
             float* __restrict__ xw, int n) {
    const int lane = threadIdx.x & 31, wave = threadIdx.x >> 5;
    const int half = lane >> 4, mrow = lane & 15;
    const int rowBase = blockIdx.x * 128 + wave * 16;
    int arow = rowBase + mrow; if (arow >= n) arow = n - 1;
    const float* ap0 = x + (unsigned)arow * IN_F + half * 8;

    v8f acc[8];
    #pragma unroll
    for (int c = 0; c < 8; ++c) acc[c] = (v8f){0,0,0,0,0,0,0,0};

    for (int kt = 0; kt < IN_F / 32; ++kt) {
        v16bf af = make_afrag<false>(ap0 + kt * 32);
        #pragma unroll
        for (int c = 0; c < 8; ++c) {
            v16bf bv = *(const v16bf*)(bf + (((c * 24 + kt) * 32) + lane) * 16);
            acc[c] = __builtin_amdgcn_wmma_f32_16x16x32_bf16(
                false, af, false, bv, (short)0, acc[c], false, false);
        }
    }
    // epilogue: 32-bit addressing; unguarded fast path for full tiles
    const unsigned obase = (unsigned)(rowBase + 8 * half) * 128u + mrow;
    if (rowBase + 16 <= n) {
        #pragma unroll
        for (int c = 0; c < 8; ++c)
            #pragma unroll
            for (int r = 0; r < 8; ++r)
                xw[obase + (unsigned)r * 128u + c * 16] = acc[c][r];
    } else {
        #pragma unroll
        for (int r = 0; r < 8; ++r)
            if (rowBase + 8 * half + r < n)
                #pragma unroll
                for (int c = 0; c < 8; ++c)
                    xw[obase + (unsigned)r * 128u + c * 16] = acc[c][r];
    }
}

// ---------------------------------------------------------------- GEMM2: xw2 = relu(agg1[:,branch]) @ W2[:64] + rootproj[batch]
__global__ __launch_bounds__(256)
void k_gemm2(const float* __restrict__ agg1, const __bf16* __restrict__ bf2,
             const float* __restrict__ rootproj, const int* __restrict__ batch,
             float* __restrict__ xw2, int n, int nB) {
    const int branch = blockIdx.y, cOff = branch * 64;
    const int lane = threadIdx.x & 31, wave = threadIdx.x >> 5;
    const int half = lane >> 4, mrow = lane & 15;
    const int rowBase = blockIdx.x * 128 + wave * 16;
    int arow = rowBase + mrow; if (arow >= n) arow = n - 1;
    const float* ap0 = agg1 + (unsigned)arow * 128u + cOff + half * 8;
    const __bf16* bfb = bf2 + branch * (4 * 2 * 32 * 16);
    const float* rp = rootproj + branch * nB * 64;

    v8f acc[4];
    #pragma unroll
    for (int c = 0; c < 4; ++c) acc[c] = (v8f){0,0,0,0,0,0,0,0};

    #pragma unroll
    for (int kt = 0; kt < 2; ++kt) {
        v16bf af = make_afrag<true>(ap0 + kt * 32);   // fused relu
        #pragma unroll
        for (int c = 0; c < 4; ++c) {
            v16bf bv = *(const v16bf*)(bfb + (((c * 2 + kt) * 32) + lane) * 16);
            acc[c] = __builtin_amdgcn_wmma_f32_16x16x32_bf16(
                false, af, false, bv, (short)0, acc[c], false, false);
        }
    }
    // hoist the 8 row->graph lookups (one per accumulator row)
    int bidx[8];
    #pragma unroll
    for (int r = 0; r < 8; ++r) {
        int orow = rowBase + 8 * half + r;
        bidx[r] = batch[orow < n ? orow : (n - 1)];
    }
    const unsigned obase = (unsigned)(rowBase + 8 * half) * 128u + cOff + mrow;
    if (rowBase + 16 <= n) {
        #pragma unroll
        for (int c = 0; c < 4; ++c) {
            int col = c * 16 + mrow;
            #pragma unroll
            for (int r = 0; r < 8; ++r)
                xw2[obase + (unsigned)r * 128u + c * 16] =
                    acc[c][r] + rp[bidx[r] * 64 + col];
        }
    } else {
        #pragma unroll
        for (int r = 0; r < 8; ++r)
            if (rowBase + 8 * half + r < n)
                #pragma unroll
                for (int c = 0; c < 4; ++c)
                    xw2[obase + (unsigned)r * 128u + c * 16] =
                        acc[c][r] + rp[bidx[r] * 64 + c * 16 + mrow];
    }
}

// ---------------------------------------------------------------- aggregation
// agg[i,j] = dinv[i]^2 * xw[i,j] + bias[j]   (self-loop term + bias)
__global__ void k_agg_init(const float* __restrict__ xw, const float* __restrict__ dtd,
                           const float* __restrict__ dbu, const float* __restrict__ btd,
                           const float* __restrict__ bbu, float* __restrict__ agg, int n) {
    unsigned t = blockIdx.x * blockDim.x + threadIdx.x;
    if (t >= (unsigned)n * 128u) return;
    unsigned i = t >> 7; int j = (int)(t & 127);
    float dv = (j < 64) ? dtd[i] : dbu[i];
    float bias = (j < 64) ? btd[j] : bbu[j - 64];
    agg[t] = dv * dv * xw[t] + bias;
}

// one wave per edge, 2 features per lane, f32 atomics (L2-resident targets)
__global__ __launch_bounds__(256)
void k_agg_edges(const int* __restrict__ src, const int* __restrict__ dst,
                 const float* __restrict__ dinv, const float* __restrict__ xw,
                 float* __restrict__ agg, int e, int cOff) {
    int w = (int)((blockIdx.x * 256u + threadIdx.x) >> 5);
    int lane = threadIdx.x & 31;
    if (w >= e) return;
    int s = src[w], d = dst[w];
    float nrm = dinv[s] * dinv[d];
    unsigned sb = (unsigned)s * 128u + cOff + lane * 2;
    unsigned db = (unsigned)d * 128u + cOff + lane * 2;
    float2 v = *(const float2*)(xw + sb);
    atomicAdd(agg + db,     nrm * v.x);
    atomicAdd(agg + db + 1, nrm * v.y);
}

// ---------------------------------------------------------------- pooling + head
__global__ void k_pool_zero(float* __restrict__ sums, float* __restrict__ cnt, int nB) {
    int i = blockIdx.x * blockDim.x + threadIdx.x;
    if (i < nB * 128) sums[i] = 0.0f;
    if (i < nB) cnt[i] = 0.0f;
}

__global__ void k_pool(const float* __restrict__ agg2, const int* __restrict__ batch,
                       float* __restrict__ sums, float* __restrict__ cnt, int n, int nB) {
    unsigned t = blockIdx.x * blockDim.x + threadIdx.x;
    if (t >= (unsigned)n * 128u) return;
    unsigned i = t >> 7; int j = (int)(t & 127);
    int b = batch[i];
    float v = fmaxf(agg2[t], 0.0f);
    atomicAdd(sums + ((j >> 6) * nB + b) * 64 + (j & 63), v);
    if (j == 0) atomicAdd(cnt + b, 1.0f);
}

// feat = [bu_mean(64), bu_root(64), td_mean(64), td_root(64)] @ fc_W + fc_b -> log_softmax
__global__ void k_head(const float* __restrict__ sums, const float* __restrict__ cnt,
                       const float* __restrict__ agg1, const int* __restrict__ rootindex,
                       const float* __restrict__ fcW, const float* __restrict__ fcb,
                       float* __restrict__ out, int nB) {
    int b = blockIdx.x * blockDim.x + threadIdx.x;
    if (b >= nB) return;
    unsigned ri = (unsigned)rootindex[b];
    float inv = 1.0f / fmaxf(cnt[b], 1.0f);
    float lg[NCLS] = {fcb[0], fcb[1], fcb[2], fcb[3]};
    for (int f = 0; f < 256; ++f) {
        float v;
        if (f < 64)       v = sums[(nB + b) * 64 + f] * inv;           // bu mean
        else if (f < 128) v = agg1[ri * 128u + 64 + (f - 64)];         // bu root (x2)
        else if (f < 192) v = sums[b * 64 + (f - 128)] * inv;          // td mean
        else              v = agg1[ri * 128u + (f - 192)];             // td root
        #pragma unroll
        for (int c = 0; c < NCLS; ++c) lg[c] += v * fcW[f * NCLS + c];
    }
    float mx = fmaxf(fmaxf(lg[0], lg[1]), fmaxf(lg[2], lg[3]));
    float sum = 0.0f;
    #pragma unroll
    for (int c = 0; c < NCLS; ++c) sum += expf(lg[c] - mx);
    float lse = logf(sum);
    #pragma unroll
    for (int c = 0; c < NCLS; ++c) out[b * NCLS + c] = lg[c] - mx - lse;
}

// ---------------------------------------------------------------- launch
extern "C" void kernel_launch(void* const* d_in, const int* in_sizes, int n_in,
                              void* d_out, int out_size, void* d_ws, size_t ws_size,
                              hipStream_t stream) {
    const float* x      = (const float*)d_in[0];
    const int*   ei     = (const int*)d_in[1];
    const int*   bei    = (const int*)d_in[2];
    const int*   batch  = (const int*)d_in[3];
    const int*   rooti  = (const int*)d_in[4];
    const float* root   = (const float*)d_in[5];
    const float* W1td   = (const float*)d_in[6];
    const float* b1td   = (const float*)d_in[7];
    const float* W2td   = (const float*)d_in[8];
    const float* b2td   = (const float*)d_in[9];
    const float* W1bu   = (const float*)d_in[10];
    const float* b1bu   = (const float*)d_in[11];
    const float* W2bu   = (const float*)d_in[12];
    const float* b2bu   = (const float*)d_in[13];
    const float* fcW    = (const float*)d_in[14];
    const float* fcb    = (const float*)d_in[15];
    float* out = (float*)d_out;

    const int n  = in_sizes[3];           // nodes
    const int e  = in_sizes[1] / 2;       // edges per branch
    const int nB = in_sizes[4];           // graphs

    // workspace carve (aligned 256B)
    char* p = (char*)d_ws;
    auto take = [&](size_t bytes) -> char* {
        char* r = p; p += (bytes + 255) & ~(size_t)255; return r;
    };
    float*  xw   = (float*)take((size_t)n * 128 * 4);  // GEMM1 out, reused as GEMM2 out
    float*  agg1 = (float*)take((size_t)n * 128 * 4);  // conv1 out (pre-relu; x2 source)
    float*  agg2 = (float*)take((size_t)n * 128 * 4);  // conv2 out (pre-relu)
    float*  dtd  = (float*)take((size_t)n * 4);
    float*  dbu  = (float*)take((size_t)n * 4);
    __bf16* bf1  = (__bf16*)take((size_t)IN_F * 128 * 2);
    __bf16* bf2  = (__bf16*)take((size_t)2 * 64 * 64 * 2);
    float*  rp   = (float*)take((size_t)2 * nB * 64 * 4);
    float*  sums = (float*)take((size_t)2 * nB * 64 * 4);
    float*  cnt  = (float*)take((size_t)nB * 4);
    (void)ws_size; (void)n_in; (void)out_size;

    const int T = 256;
    long tot = (long)n * 128;
    int nb_n    = (n + T - 1) / T;
    int nb_e    = (e + T - 1) / T;
    int nb_tot  = (int)((tot + T - 1) / T);
    int nb_ew   = (e + 7) / 8;              // wave per edge
    int nb_gemm = (n + 127) / 128;

    // degrees -> dinv (both branches)
    k_init_deg<<<nb_n, T, 0, stream>>>(dtd, dbu, n);
    k_deg_edges<<<nb_e, T, 0, stream>>>(ei + e, dtd, e);
    k_deg_edges<<<nb_e, T, 0, stream>>>(bei + e, dbu, e);
    k_dinv<<<nb_n, T, 0, stream>>>(dtd, dbu, n);

    // pack weights into WMMA B-fragment layout; root projection
    k_pack_b1<<<(8 * 24 * 32 * 16 + T - 1) / T, T, 0, stream>>>(W1td, W1bu, bf1);
    k_pack_b2<<<(2 * 4 * 2 * 32 * 16 + T - 1) / T, T, 0, stream>>>(W2td, W2bu, bf2);
    k_rootproj<<<dim3(nB, 2), 64, 0, stream>>>(root, W2td, W2bu, rp, nB);

    // conv1: fused both-branch GEMM, then self-loop init + edge scatter
    k_gemm1<<<nb_gemm, T, 0, stream>>>(x, bf1, xw, n);
    k_agg_init<<<nb_tot, T, 0, stream>>>(xw, dtd, dbu, b1td, b1bu, agg1, n);
    k_agg_edges<<<nb_ew, T, 0, stream>>>(ei, ei + e, dtd, xw, agg1, e, 0);
    k_agg_edges<<<nb_ew, T, 0, stream>>>(bei, bei + e, dbu, xw, agg1, e, 64);

    // conv2: relu fused into A-load, root projection fused into epilogue
    k_gemm2<<<dim3(nb_gemm, 2), T, 0, stream>>>(agg1, bf2, rp, batch, xw, n, nB);
    k_agg_init<<<nb_tot, T, 0, stream>>>(xw, dtd, dbu, b2td, b2bu, agg2, n);
    k_agg_edges<<<nb_ew, T, 0, stream>>>(ei, ei + e, dtd, xw, agg2, e, 0);
    k_agg_edges<<<nb_ew, T, 0, stream>>>(bei, bei + e, dbu, xw, agg2, e, 64);

    // mean pooling + classifier head
    k_pool_zero<<<(nB * 128 + T - 1) / T, T, 0, stream>>>(sums, cnt, nB);
    k_pool<<<nb_tot, T, 0, stream>>>(agg2, batch, sums, cnt, n, nB);
    k_head<<<(nB + 127) / 128, 128, 0, stream>>>(sums, cnt, agg1, rooti, fcW, fcb, out, nB);
}
